// MDRNN_11364483465546
// MI455X (gfx1250) — matrix operand order repaired and checked by reading, not compiled, not run on hardware
//
#include <hip/hip_runtime.h>
#include <hip/hip_bf16.h>

#define S_   64
#define B_   128
#define A_   32
#define L_   128
#define H_   512
#define H4_  2048
#define KC_  160      // L + A
#define SB_  8192     // S*B
#define NG_  2058     // GMM_OUT
#define NGP_ 2112     // padded to multiple of 64 (4 n-tiles/wave * 16)

typedef __attribute__((ext_vector_type(8)))  float          fx8;
typedef __attribute__((ext_vector_type(16))) __bf16         bfx16;
typedef __attribute__((ext_vector_type(8)))  unsigned short usx8;
typedef __attribute__((ext_vector_type(16))) unsigned short usx16;

union UFragA { usx8  h[2]; bfx16 b; };
union UFragB { usx16 u;    bfx16 b; };

__device__ __forceinline__ unsigned short f2bf(float f) {
    unsigned int u = __float_as_uint(f);
    u += 0x7fffu + ((u >> 16) & 1u);       // round-to-nearest-even
    return (unsigned short)(u >> 16);
}
__device__ __forceinline__ float sigf(float x) { return 1.0f / (1.0f + __expf(-x)); }

// ---- CDNA5 async Global->LDS copy (ASYNCcnt-tracked) ------------------------
__device__ __forceinline__ void async_ld_b128(unsigned short* sdst /*generic, ->LDS*/,
                                              const unsigned short* gsrc)
{
    unsigned int loff = (unsigned int)(unsigned long long)
        (__attribute__((address_space(3))) unsigned short*)sdst;
    asm volatile("global_load_async_to_lds_b128 %0, %1, off"
                 :: "v"(loff), "v"((const void*)gsrc) : "memory");
}
__device__ __forceinline__ void wait_async0()
{
#if __has_builtin(__builtin_amdgcn_s_wait_asynccnt)
    __builtin_amdgcn_s_wait_asynccnt(0);
#else
    asm volatile("s_wait_asynccnt 0x0" ::: "memory");
#endif
}

// ---------------------------------------------------------------------------
// WMMA GEMM: C[M,N] = A[M,K](bf16 row-major, row stride lda)
//                   @ Bp (bf16, fragment-packed (K/32, N/16, 32 lanes, 16 elems))
//                   + addend(f32, MxN, opt) + bias(f32, len Nbias, opt)
// block = 256 thr (8 waves) -> 128 rows x 64 cols; wave w: rows m0..m0+15, 4 n-tiles.
// B chunk (4 tiles, 4KB) staged in LDS via async copy, double buffered.
// grid (N/64, M/128)
// ---------------------------------------------------------------------------
__global__ __launch_bounds__(256)
void k_gemm_bf16(const unsigned short* __restrict__ A,
                 const unsigned short* __restrict__ Bp,
                 float* __restrict__ C,
                 const float* __restrict__ addend,
                 const float* __restrict__ bias,
                 int M, int N, int K, int lda, int Nbias)
{
    __shared__ __align__(16) unsigned short smem[2][2048];   // 2 x 4KB

    const int tid  = threadIdx.x;
    const int wave = tid >> 5;
    const int lane = tid & 31;
    const int m0   = blockIdx.y * 128 + wave * 16;
    const int m    = m0 + (lane & 15);
    const int g    = lane >> 4;
    const int ntn  = N >> 4;                 // 16-wide tiles in N
    const int tn0  = blockIdx.x * 4;         // first of this block's 4 n-tiles
    const int KT   = K >> 5;

    const unsigned short* arow = A + (long)m * lda;
    // per-K-tile source chunk: 4 consecutive packed tiles = 2048 ushorts
    const unsigned short* bsrc0 = Bp + ((long)0 * ntn + tn0) * 512 + tid * 8;
    const long bstep = (long)ntn * 512;      // advance one K-tile

    fx8 acc0 = {}, acc1 = {}, acc2 = {}, acc3 = {};

    // prologue: stage K-tile 0 into buffer 0
    async_ld_b128(&smem[0][tid * 8], bsrc0);

    for (int tk = 0; tk < KT; ++tk) {
        const int cur = tk & 1;
        wait_async0();          // my copy of smem[cur] complete
        __syncthreads();        // all waves' copies complete; prev reads consumed
        if (tk + 1 < KT)
            async_ld_b128(&smem[cur ^ 1][tid * 8], bsrc0 + (long)(tk + 1) * bstep);

        const int kk = tk << 5;
        UFragA au;
        au.h[0] = *(const usx8*)(arow + kk + g * 16);
        au.h[1] = *(const usx8*)(arow + kk + g * 16 + 8);
        if (tk + 1 < KT)
            __builtin_prefetch(arow + kk + 32, 0, 1);

        UFragB b0, b1, b2, b3;
        b0.u = *(const usx16*)(&smem[cur][0 * 512 + lane * 16]);
        b1.u = *(const usx16*)(&smem[cur][1 * 512 + lane * 16]);
        b2.u = *(const usx16*)(&smem[cur][2 * 512 + lane * 16]);
        b3.u = *(const usx16*)(&smem[cur][3 * 512 + lane * 16]);

        acc0 = __builtin_amdgcn_wmma_f32_16x16x32_bf16(false, au.b, false, b0.b, (short)0, acc0, false, false);
        acc1 = __builtin_amdgcn_wmma_f32_16x16x32_bf16(false, au.b, false, b1.b, (short)0, acc1, false, false);
        acc2 = __builtin_amdgcn_wmma_f32_16x16x32_bf16(false, au.b, false, b2.b, (short)0, acc2, false, false);
        acc3 = __builtin_amdgcn_wmma_f32_16x16x32_bf16(false, au.b, false, b3.b, (short)0, acc3, false, false);
    }

    fx8 accs[4] = { acc0, acc1, acc2, acc3 };
#pragma unroll
    for (int j = 0; j < 4; ++j) {
        const int n = (tn0 + j) * 16 + (lane & 15);
#pragma unroll
        for (int q = 0; q < 8; ++q) {
            const int r = m0 + g * 8 + q;
            float v = accs[j][q];
            if (addend) v += addend[(long)r * N + n];
            if (bias && n < Nbias) v += bias[n];
            C[(long)r * N + n] = v;
        }
    }
}

// ---------------------------------------------------------------------------
// Pack weight W[Nreal, ldW] (f32, column slice at k_off) into WMMA-B fragment
// order (bf16): dest[((tk*Npad/16+tn)*32+lane)*16+j] = W[n, k+k_off],
// n = tn*16+(lane&15), k = tk*32+(lane>>4)*16+j ; pads n>=Nreal with 0.
// ---------------------------------------------------------------------------
__global__ void k_pack_b(const float* __restrict__ W, unsigned short* __restrict__ Bp,
                         int K, int Npad, int Nreal, int ldW, int k_off)
{
    int idx = blockIdx.x * blockDim.x + threadIdx.x;
    if (idx >= K * Npad) return;
    int j    = idx & 15;
    int lane = (idx >> 4) & 31;
    int tile = idx >> 9;
    int ntn  = Npad >> 4;
    int tn   = tile % ntn;
    int tk   = tile / ntn;
    int n    = tn * 16 + (lane & 15);
    int k    = tk * 32 + (lane >> 4) * 16 + j;
    float v  = (n < Nreal) ? W[(long)n * ldW + k + k_off] : 0.0f;
    Bp[idx]  = f2bf(v);
}

__global__ void k_addv(const float* a, const float* b, float* o, int n) {
    int i = blockIdx.x * blockDim.x + threadIdx.x;
    if (i < n) o[i] = a[i] + b[i];
}

// concat [actions(S,B,32), latents(S,B,128)] -> bf16 (S*B,160)
__global__ void k_concat_ins(const float* __restrict__ act, const float* __restrict__ lat,
                             unsigned short* __restrict__ ins)
{
    int idx = blockIdx.x * blockDim.x + threadIdx.x;
    if (idx >= SB_ * KC_) return;
    int row = idx / KC_, j = idx % KC_;
    float v = (j < A_) ? act[row * A_ + j] : lat[row * L_ + (j - A_)];
    ins[idx] = f2bf(v);
}

__global__ void k_f32_to_bf(const float* __restrict__ src, unsigned short* __restrict__ dst,
                            int rows, int cols, int sstride, int dstride)
{
    int idx = blockIdx.x * blockDim.x + threadIdx.x;
    if (idx >= rows * cols) return;
    int r = idx / cols, c = idx % cols;
    dst[r * dstride + c] = f2bf(src[r * sstride + c]);
}

// z(B,4H) gate order i,f,g,o ; updates h,c (f32); optional bf16 h mirror / outs
__global__ void k_lstm_gates(const float* __restrict__ z,
                             float* __restrict__ h, float* __restrict__ c,
                             unsigned short* __restrict__ h_bf, int hbf_stride,
                             float* __restrict__ out_f, unsigned short* __restrict__ out_bf)
{
    int idx = blockIdx.x * blockDim.x + threadIdx.x;
    if (idx >= B_ * H_) return;
    int b = idx / H_, j = idx % H_;
    const float* zr = z + (long)b * H4_;
    float ig = sigf(zr[j]);
    float fg = sigf(zr[H_ + j]);
    float gg = tanhf(zr[2 * H_ + j]);
    float og = sigf(zr[3 * H_ + j]);
    float cn = fg * c[idx] + ig * gg;
    float hn = og * tanhf(cn);
    c[idx] = cn;
    h[idx] = hn;
    if (h_bf)   h_bf[b * hbf_stride + j] = f2bf(hn);
    if (out_f)  out_f[idx]  = hn;
    if (out_bf) out_bf[idx] = f2bf(hn);
}

// scores[s*B+b] = relu(hproj[b,:] + outs_proj[s*B+b,:]) . W2 + b2 ; one wave per (s,b)
__global__ __launch_bounds__(256)
void k_attn_scores(const float* __restrict__ hproj, const float* __restrict__ outs_proj,
                   const float* __restrict__ W2, const float* __restrict__ b2,
                   float* __restrict__ scores)
{
    int wave = threadIdx.x >> 5, lane = threadIdx.x & 31;
    int p = blockIdx.x * 8 + wave;             // p = s*B + b
    int b = p & (B_ - 1);
    const float* hp = hproj + (long)b * H_;
    const float* op = outs_proj + (long)p * H_;
    float acc = 0.0f;
    for (int j = lane; j < H_; j += 32) {
        float v = hp[j] + op[j];
        v = v > 0.0f ? v : 0.0f;
        acc += v * W2[j];
    }
#pragma unroll
    for (int off = 16; off; off >>= 1) acc += __shfl_xor(acc, off, 32);
    if (lane == 0) scores[p] = acc + b2[0];
}

// softmax over the BATCH axis (reference quirk): per s, over the 128 b's
__global__ void k_attn_softmax(const float* __restrict__ scores, float* __restrict__ wts)
{
    __shared__ float red[B_];
    int s = blockIdx.x, b = threadIdx.x;
    float x = scores[s * B_ + b];
    red[b] = x; __syncthreads();
    for (int off = B_ / 2; off; off >>= 1) {
        if (b < off) red[b] = fmaxf(red[b], red[b + off]);
        __syncthreads();
    }
    float mx = red[0]; __syncthreads();
    float e = __expf(x - mx);
    red[b] = e; __syncthreads();
    for (int off = B_ / 2; off; off >>= 1) {
        if (b < off) red[b] += red[b + off];
        __syncthreads();
    }
    wts[s * B_ + b] = e / red[0];
}

// context[b,j] = sum_s wts[s*B+b]*outs[(s*B+b)*H+j] -> bf16 into cat buffer (stride 2H)
__global__ void k_attn_context(const float* __restrict__ wts, const float* __restrict__ outs,
                               unsigned short* __restrict__ cat_bf)
{
    int idx = blockIdx.x * blockDim.x + threadIdx.x;
    if (idx >= B_ * H_) return;
    int b = idx >> 9, j = idx & (H_ - 1);
    float acc = 0.0f;
    for (int s = 0; s < S_; ++s)
        acc += wts[s * B_ + b] * outs[((long)(s * B_ + b)) * H_ + j];
    cat_bf[b * (2 * H_) + j] = f2bf(acc);
}

// gmm(SB, NGP) -> mus, sigmas=exp, logpi=log_softmax(8), rs, ds ; one wave per row
__global__ __launch_bounds__(256)
void k_gmm_out(const float* __restrict__ gmm,
               float* __restrict__ mus, float* __restrict__ sig,
               float* __restrict__ logpi, float* __restrict__ rs, float* __restrict__ ds)
{
    int wave = threadIdx.x >> 5, lane = threadIdx.x & 31;
    int row = blockIdx.x * 8 + wave;
    if (row >= SB_) return;
    const float* gr = gmm + (long)row * NGP_;
    for (int j = lane; j < 1024; j += 32) {
        mus[(long)row * 1024 + j] = gr[j];
        sig[(long)row * 1024 + j] = __expf(gr[1024 + j]);
    }
    if (lane == 0) {
        float x[8], mx = -3.4e38f;
#pragma unroll
        for (int g = 0; g < 8; ++g) { x[g] = gr[2048 + g]; mx = fmaxf(mx, x[g]); }
        float sum = 0.0f;
#pragma unroll
        for (int g = 0; g < 8; ++g) sum += __expf(x[g] - mx);
        float lse = mx + __logf(sum);
#pragma unroll
        for (int g = 0; g < 8; ++g) logpi[row * 8 + g] = x[g] - lse;
        rs[row] = gr[2056];
        ds[row] = gr[2057];
    }
}

// ---------------------------------------------------------------------------
extern "C" void kernel_launch(void* const* d_in, const int* in_sizes, int n_in,
                              void* d_out, int out_size, void* d_ws, size_t ws_size,
                              hipStream_t stream)
{
    const float* actions = (const float*)d_in[0];
    const float* latents = (const float*)d_in[1];
    // d_in[2] = outputs (unused zeros)
    const float* Wih_e = (const float*)d_in[3];
    const float* Whh_e = (const float*)d_in[4];
    const float* bih_e = (const float*)d_in[5];
    const float* bhh_e = (const float*)d_in[6];
    const float* Wih_d = (const float*)d_in[7];
    const float* Whh_d = (const float*)d_in[8];
    const float* bih_d = (const float*)d_in[9];
    const float* bhh_d = (const float*)d_in[10];
    const float* W1    = (const float*)d_in[11];
    const float* b1    = (const float*)d_in[12];
    const float* W2    = (const float*)d_in[13];
    const float* b2    = (const float*)d_in[14];
    const float* Wg    = (const float*)d_in[15];
    const float* bg    = (const float*)d_in[16];

    // ---- workspace carve-out ------------------------------------------------
    char* base = (char*)d_ws;
    size_t off = 0;
    auto alloc = [&](size_t bytes) -> char* {
        off = (off + 255) & ~(size_t)255;
        char* p = base + off;
        off += bytes;
        return p;
    };
    float* Xih       = (float*)alloc((size_t)SB_ * H4_ * 4);
    float* outs      = (float*)alloc((size_t)SB_ * H_ * 4);
    float* outs_proj = (float*)alloc((size_t)SB_ * H_ * 4);
    float* gmm       = (float*)alloc((size_t)SB_ * NGP_ * 4);
    float* zbuf      = (float*)alloc((size_t)B_ * H4_ * 4);
    float* hst       = (float*)alloc((size_t)B_ * H_ * 4);
    float* cst       = (float*)alloc((size_t)B_ * H_ * 4);
    float* hproj     = (float*)alloc((size_t)B_ * H_ * 4);
    float* scores    = (float*)alloc((size_t)S_ * B_ * 4);
    float* wts       = (float*)alloc((size_t)S_ * B_ * 4);
    float* b_e       = (float*)alloc((size_t)H4_ * 4);
    float* b_d       = (float*)alloc((size_t)H4_ * 4);
    unsigned short* ins_bf      = (unsigned short*)alloc((size_t)SB_ * KC_ * 2);
    unsigned short* h_bf_enc    = (unsigned short*)alloc((size_t)B_ * H_ * 2);
    unsigned short* cat_bf      = (unsigned short*)alloc((size_t)B_ * 2 * H_ * 2);
    unsigned short* outs_bf     = (unsigned short*)alloc((size_t)SB_ * H_ * 2);
    unsigned short* dec_outs_bf = (unsigned short*)alloc((size_t)SB_ * H_ * 2);
    unsigned short* WihE_p = (unsigned short*)alloc((size_t)KC_ * H4_ * 2);
    unsigned short* WhhE_p = (unsigned short*)alloc((size_t)H_ * H4_ * 2);
    unsigned short* W1a_p  = (unsigned short*)alloc((size_t)H_ * H_ * 2);
    unsigned short* W1b_p  = (unsigned short*)alloc((size_t)H_ * H_ * 2);
    unsigned short* Wd_p   = (unsigned short*)alloc((size_t)2 * H_ * H4_ * 2);
    unsigned short* Wg_p   = (unsigned short*)alloc((size_t)H_ * NGP_ * 2);

    float* out_mus   = (float*)d_out;
    float* out_sig   = out_mus + (size_t)SB_ * 1024;
    float* out_logpi = out_sig + (size_t)SB_ * 1024;
    float* out_rs    = out_logpi + (size_t)SB_ * 8;
    float* out_ds    = out_rs + SB_;

    auto gemm = [&](const unsigned short* A, const unsigned short* Bp, float* C,
                    const float* addend, const float* bias,
                    int M, int N, int K, int lda, int Nbias) {
        dim3 grid(N / 64, M / 128);
        k_gemm_bf16<<<grid, 256, 0, stream>>>(A, Bp, C, addend, bias, M, N, K, lda, Nbias);
    };
    auto pack = [&](const float* W, unsigned short* Bp, int K, int Npad, int Nreal,
                    int ldW, int koff) {
        int total = K * Npad;
        k_pack_b<<<(total + 255) / 256, 256, 0, stream>>>(W, Bp, K, Npad, Nreal, ldW, koff);
    };

    // ---- prep (deterministic, every call) -----------------------------------
    hipMemsetAsync(hst, 0, (size_t)B_ * H_ * 4, stream);
    hipMemsetAsync(cst, 0, (size_t)B_ * H_ * 4, stream);
    hipMemsetAsync(h_bf_enc, 0, (size_t)B_ * H_ * 2, stream);

    k_addv<<<(H4_ + 255) / 256, 256, 0, stream>>>(bih_e, bhh_e, b_e, H4_);
    k_addv<<<(H4_ + 255) / 256, 256, 0, stream>>>(bih_d, bhh_d, b_d, H4_);

    pack(Wih_e, WihE_p, KC_, H4_, H4_, KC_, 0);
    pack(Whh_e, WhhE_p, H_, H4_, H4_, H_, 0);
    pack(W1, W1a_p, H_, H_, H_, 2 * H_, 0);        // W1[:, :H]
    pack(W1, W1b_p, H_, H_, H_, 2 * H_, H_);       // W1[:, H:]
    pack(Wih_d, Wd_p, H_, H4_, H4_, H_, 0);        // K rows 0..511
    pack(Whh_d, Wd_p + (size_t)H_ * H4_, H_, H4_, H4_, H_, 0); // K rows 512..1023
    pack(Wg, Wg_p, H_, NGP_, NG_, H_, 0);

    k_concat_ins<<<(SB_ * KC_ + 255) / 256, 256, 0, stream>>>(actions, latents, ins_bf);

    // X projection for all encoder steps: (8192,160)@(160,2048)
    gemm(ins_bf, WihE_p, Xih, nullptr, nullptr, SB_, H4_, KC_, KC_, 0);

    // ---- encoder LSTM: 64 sequential steps ----------------------------------
    for (int s = 0; s < S_; ++s) {
        gemm(h_bf_enc, WhhE_p, zbuf, Xih + (size_t)s * B_ * H4_, b_e,
             B_, H4_, H_, H_, H4_);
        k_lstm_gates<<<(B_ * H_ + 255) / 256, 256, 0, stream>>>(
            zbuf, hst, cst, h_bf_enc, H_,
            outs + (size_t)s * B_ * H_, outs_bf + (size_t)s * B_ * H_);
    }

    // precompute outs @ W1[:,H:]^T + b1  (constant across decoder steps)
    gemm(outs_bf, W1b_p, outs_proj, nullptr, b1, SB_, H_, H_, H_, H_);
    // seed decoder h_bf (= hT) into cat buffer second half (stride 2H)
    k_f32_to_bf<<<(B_ * H_ + 255) / 256, 256, 0, stream>>>(
        hst, cat_bf + H_, B_, H_, H_, 2 * H_);

    // ---- decoder: 64 sequential steps ---------------------------------------
    for (int t = 0; t < S_; ++t) {
        // h @ W1[:, :H]^T
        gemm(cat_bf + H_, W1a_p, hproj, nullptr, nullptr, B_, H_, H_, 2 * H_, 0);
        k_attn_scores<<<SB_ / 8, 256, 0, stream>>>(hproj, outs_proj, W2, b2, scores);
        k_attn_softmax<<<S_, B_, 0, stream>>>(scores, wts);
        k_attn_context<<<(B_ * H_ + 255) / 256, 256, 0, stream>>>(wts, outs, cat_bf);
        // fused [context,h] @ [Wih_d;Whh_d]^T + b_d  (K = 1024)
        gemm(cat_bf, Wd_p, zbuf, nullptr, b_d, B_, H4_, 2 * H_, 2 * H_, H4_);
        k_lstm_gates<<<(B_ * H_ + 255) / 256, 256, 0, stream>>>(
            zbuf, hst, cst, cat_bf + H_, 2 * H_,
            nullptr, dec_outs_bf + (size_t)t * B_ * H_);
    }

    // ---- GMM head -----------------------------------------------------------
    gemm(dec_outs_bf, Wg_p, gmm, nullptr, bg, SB_, NGP_, H_, H_, NG_);
    k_gmm_out<<<SB_ / 8, 256, 0, stream>>>(gmm, out_mus, out_sig, out_logpi, out_rs, out_ds);
}
